// VertPosNet_58042188038825
// MI455X (gfx1250) — matrix operand order repaired and checked by reading, not compiled, not run on hardware
//
#include <hip/hip_runtime.h>
#include <hip/hip_bf16.h>

typedef __attribute__((ext_vector_type(16))) __bf16 v16bf;
typedef __attribute__((ext_vector_type(2)))  __bf16 bf16x2;
typedef __attribute__((ext_vector_type(8)))  float  v8f;

#define N_NODES    100000
#define N_EDGES    300000
#define N_BOUNDARY 1024

// ---------------------------------------------------------------- zero
__global__ void zero_kernel(float* __restrict__ p, int n) {
    int i = blockIdx.x * blockDim.x + threadIdx.x;
    if (i < n) p[i] = 0.0f;
}

// ---------------------------------------------------------------- edge scatter
// agg[dst] += x[src]; agg[src] += x[dst];  4 channels per thread (or C if C<4).
__global__ void scatter_edges(const float* __restrict__ x,
                              const int*   __restrict__ edges,
                              float*       __restrict__ agg,
                              int n_edges, int C, int chunks, int cpt) {
    int tid = blockIdx.x * blockDim.x + threadIdx.x;
    int e  = tid / chunks;
    if (e >= n_edges) return;
    int c0 = (tid % chunks) * cpt;
    int src = edges[2 * e + 0];
    int dst = edges[2 * e + 1];
    int cend = c0 + cpt; if (cend > C) cend = C;
    for (int c = c0; c < cend; ++c) {
        float xs = x[(size_t)src * C + c];
        float xd = x[(size_t)dst * C + c];
        atomicAdd(&agg[(size_t)dst * C + c], xs);
        atomicAdd(&agg[(size_t)src * C + c], xd);
    }
}

// ---------------------------------------------------------------- layer 0 (K=3, VALU)
__global__ void conv0_kernel(const float* __restrict__ x,
                             const float* __restrict__ agg,
                             const float* __restrict__ Ws,   // 3 x 64
                             const float* __restrict__ Wn,   // 3 x 64
                             const float* __restrict__ bias, // 64
                             float*       __restrict__ y,    // M x 64
                             int M) {
    int tid = blockIdx.x * blockDim.x + threadIdx.x;
    int m = tid >> 6;
    int n = tid & 63;
    if (m >= M) return;
    float acc = bias[n];
    #pragma unroll
    for (int k = 0; k < 3; ++k) {
        acc += x[(size_t)m * 3 + k]   * Ws[k * 64 + n];
        acc += agg[(size_t)m * 3 + k] * Wn[k * 64 + n];
    }
    y[(size_t)m * 64 + n] = acc > 0.0f ? acc : 0.0f;
}

// ---------------------------------------------------------------- pack activations into WMMA A-fragment order
// Fragment f = mt*(K/32)+k32 ; 32 lanes x 16 bf16, stored contiguously (32B/lane).
// A 16-bit 16x32 ISA layout: lane m=lane%16, half=lane/16; VGPR v<4 -> K=8h+2v, v>=4 -> 16+8h+2(v-4).
// One thread per bf16 pair.
__global__ void pack_A_frag(const float* __restrict__ X, __bf16* __restrict__ out,
                            int M, int K) {
    const int kb = K >> 5;
    size_t tid = (size_t)blockIdx.x * blockDim.x + threadIdx.x;
    size_t total = (size_t)(M >> 4) * kb * 32 * 8;
    if (tid >= total) return;
    int v    = (int)(tid & 7);
    int lane = (int)((tid >> 3) & 31);
    size_t f = tid >> 8;
    int k32 = (int)(f % kb);
    int mt  = (int)(f / kb);
    int mrow = lane & 15, half = lane >> 4;
    int k = k32 * 32 + ((v < 4) ? (half * 8 + 2 * v) : (16 + half * 8 + 2 * (v - 4)));
    const float* p = X + (size_t)(mt * 16 + mrow) * K + k;
    bf16x2 o; o.x = (__bf16)p[0]; o.y = (__bf16)p[1];
    ((bf16x2*)out)[tid] = o;
}

// ---------------------------------------------------------------- pack weights into WMMA B-fragment order
// Fragment f = nt*(K/32)+k32. B 16-bit 32x16 layout: lane n=lane%16, half=lane/16;
// VGPR r<4 -> K = 8*half + 2r, 2r+1; r>=4 -> K = 16 + 8*half + 2(r-4), +1 ; col = nt*16+n.
__global__ void pack_B_frag(const float* __restrict__ W, __bf16* __restrict__ out,
                            int K, int N) {
    const int kb = K >> 5;
    size_t tid = (size_t)blockIdx.x * blockDim.x + threadIdx.x;
    size_t total = (size_t)(N >> 4) * kb * 32 * 8;
    if (tid >= total) return;
    int r    = (int)(tid & 7);
    int lane = (int)((tid >> 3) & 31);
    size_t f = tid >> 8;
    int k32 = (int)(f % kb);
    int nt  = (int)(f / kb);
    int n = lane & 15, half = lane >> 4;
    int k = k32 * 32 + ((r < 4) ? (half * 8 + 2 * r) : (16 + half * 8 + 2 * (r - 4)));
    int col = nt * 16 + n;
    bf16x2 o;
    o.x = (__bf16)W[(size_t)k * N + col];
    o.y = (__bf16)W[(size_t)(k + 1) * N + col];
    ((bf16x2*)out)[tid] = o;
}

// ---------------------------------------------------------------- fused dual-GEMM + bias + relu on packed fragments
// Y = relu(X @ Ws + G @ Wn + b). One wave computes a 16x64 tile (4 N-tiles),
// reusing A fragments across 4 WMMA column tiles. 8 WMMAs per K-step.
__global__ void wmma_gemm_packed(const __bf16* __restrict__ Axp,
                                 const __bf16* __restrict__ Agp,
                                 const __bf16* __restrict__ Bsp,
                                 const __bf16* __restrict__ Bnp,
                                 const float*  __restrict__ bias,
                                 float*        __restrict__ Y,
                                 int M, int K, int N) {
    constexpr int NTB = 4;                    // N-tiles per wave
    const int lane = threadIdx.x & 31;
    const int wave = threadIdx.x >> 5;
    const int wpb  = blockDim.x >> 5;
    const int wid  = blockIdx.x * wpb + wave;

    const int kb      = K >> 5;
    const int ngroups = N >> 6;               // N / (16*NTB)
    const int total   = (M >> 4) * ngroups;
    if (wid >= total) return;                 // wave-uniform: EXEC stays all-ones

    const int mt = wid / ngroups;
    const int ng = wid % ngroups;

    const v16bf* Ax = (const v16bf*)Axp;
    const v16bf* Ag = (const v16bf*)Agp;
    const v16bf* Bs = (const v16bf*)Bsp;
    const v16bf* Bn = (const v16bf*)Bnp;

    v8f c[NTB] = {};
    for (int k32 = 0; k32 < kb; ++k32) {
        v16bf ax = Ax[((size_t)mt * kb + k32) * 32 + lane];
        v16bf ag = Ag[((size_t)mt * kb + k32) * 32 + lane];
        #pragma unroll
        for (int t = 0; t < NTB; ++t) {
            int nt = ng * NTB + t;
            v16bf bs = Bs[((size_t)nt * kb + k32) * 32 + lane];
            v16bf bn = Bn[((size_t)nt * kb + k32) * 32 + lane];
            c[t] = __builtin_amdgcn_wmma_f32_16x16x32_bf16(false, ax, false, bs, (short)0, c[t], false, false);
            c[t] = __builtin_amdgcn_wmma_f32_16x16x32_bf16(false, ag, false, bn, (short)0, c[t], false, false);
        }
    }

    const int ncol = lane & 15, half = lane >> 4;
    #pragma unroll
    for (int t = 0; t < NTB; ++t) {
        int col = (ng * NTB + t) * 16 + ncol;
        float bv = bias[col];
        #pragma unroll
        for (int r = 0; r < 8; ++r) {
            int row = mt * 16 + r + 8 * half;  // C/D layout: VGPR r -> M=r (lanes 0-15), M=r+8 (16-31)
            float vv = c[t][r] + bv;
            Y[(size_t)row * N + col] = vv > 0.0f ? vv : 0.0f;
        }
    }
}

// ---------------------------------------------------------------- head: boundary pool + MLP + boundary-mean add
__global__ void head_kernel(const float* __restrict__ vs,     // N_NODES x 3
                            const int*   __restrict__ bidx,   // 1024
                            const float* __restrict__ x3,     // N_NODES x 256
                            const float* __restrict__ Wd,     // 256 x 256
                            const float* __restrict__ bd,     // 256
                            const float* __restrict__ Wo,     // 256 x 12
                            const float* __restrict__ bo,     // 12
                            float*       __restrict__ out) {  // 12
    __shared__ float pooled[256];
    __shared__ float hh[256];
    __shared__ float bm[3];
    const int t = threadIdx.x;  // 256 threads

    if (t < 3) {
        float s = 0.0f;
        for (int i = 0; i < N_BOUNDARY; ++i) s += vs[(size_t)bidx[i] * 3 + t];
        bm[t] = s * (1.0f / N_BOUNDARY);
    }
    {
        float s = 0.0f;
        for (int i = 0; i < N_BOUNDARY; ++i) s += x3[(size_t)bidx[i] * 256 + t];
        pooled[t] = s * (1.0f / N_BOUNDARY);
    }
    __syncthreads();
    {
        float acc = bd[t];
        for (int k = 0; k < 256; ++k) acc += pooled[k] * Wd[k * 256 + t];
        hh[t] = acc > 0.0f ? acc : 0.0f;
    }
    __syncthreads();
    if (t < 12) {
        float acc = bo[t];
        for (int k = 0; k < 256; ++k) acc += hh[k] * Wo[k * 12 + t];
        out[t] = bm[t % 3] + acc;
    }
}

// ---------------------------------------------------------------- launch
extern "C" void kernel_launch(void* const* d_in, const int* in_sizes, int n_in,
                              void* d_out, int out_size, void* d_ws, size_t ws_size,
                              hipStream_t stream) {
    const float* vs    = (const float*)d_in[0];
    const int*   edges = (const int*)  d_in[1];
    const int*   bidx  = (const int*)  d_in[2];
    // d_in[3] = n_verts (scalar, fixed at 4)
    const float* Ws0 = (const float*)d_in[4];
    const float* Wn0 = (const float*)d_in[5];
    const float* b0  = (const float*)d_in[6];
    const float* Ws1 = (const float*)d_in[7];
    const float* Wn1 = (const float*)d_in[8];
    const float* b1  = (const float*)d_in[9];
    const float* Ws2 = (const float*)d_in[10];
    const float* Wn2 = (const float*)d_in[11];
    const float* b2  = (const float*)d_in[12];
    const float* Wd  = (const float*)d_in[13];
    const float* bd  = (const float*)d_in[14];
    const float* Wo  = (const float*)d_in[15];
    const float* bo  = (const float*)d_in[16];
    float* out = (float*)d_out;

    // workspace layout (all offsets stay 32B-aligned)
    float* ws  = (float*)d_ws;
    float* x1  = ws;                               // 100000 x 64  f32
    float* x2  = x1 + (size_t)N_NODES * 64;        // 100000 x 128 f32
    float* x3  = x2 + (size_t)N_NODES * 128;       // 100000 x 256 f32
    float* agg = x3 + (size_t)N_NODES * 256;       // up to 100000 x 128 f32
    __bf16* Axp = (__bf16*)(agg + (size_t)N_NODES * 128);   // M*K bf16 (max 12.8M)
    __bf16* Agp = Axp + (size_t)N_NODES * 128;              // M*K bf16
    __bf16* Bsp = Agp + (size_t)N_NODES * 128;              // K*N bf16 (max 32768)
    __bf16* Bnp = Bsp + 32768;

    const int TB = 256;
    auto blocks = [](long long n, int tb) { return (int)((n + tb - 1) / tb); };

    // ---- layer 0: C_in = 3 (VALU, tiny K) ----
    {
        int n = N_NODES * 3;
        zero_kernel<<<blocks(n, TB), TB, 0, stream>>>(agg, n);
        scatter_edges<<<blocks(N_EDGES, TB), TB, 0, stream>>>(vs, edges, agg, N_EDGES, 3, 1, 3);
        long long tc = (long long)N_NODES * 64;
        conv0_kernel<<<blocks(tc, TB), TB, 0, stream>>>(vs, agg, Ws0, Wn0, b0, x1, N_NODES);
    }
    // ---- layer 1: 64 -> 128, WMMA ----
    {
        const int K = 64, N = 128;
        int n = N_NODES * K;
        zero_kernel<<<blocks(n, TB), TB, 0, stream>>>(agg, n);
        long long sthreads = (long long)N_EDGES * (K / 4);
        scatter_edges<<<blocks(sthreads, TB), TB, 0, stream>>>(x1, edges, agg, N_EDGES, K, K / 4, 4);
        long long ap = (long long)N_NODES * K / 2;       // pair threads
        pack_A_frag<<<blocks(ap, TB), TB, 0, stream>>>(x1,  Axp, N_NODES, K);
        pack_A_frag<<<blocks(ap, TB), TB, 0, stream>>>(agg, Agp, N_NODES, K);
        long long bp = (long long)K * N / 2;
        pack_B_frag<<<blocks(bp, TB), TB, 0, stream>>>(Ws1, Bsp, K, N);
        pack_B_frag<<<blocks(bp, TB), TB, 0, stream>>>(Wn1, Bnp, K, N);
        int waves = (N_NODES / 16) * (N / 64);           // 12500 waves
        wmma_gemm_packed<<<blocks((long long)waves * 32, TB), TB, 0, stream>>>(
            Axp, Agp, Bsp, Bnp, b1, x2, N_NODES, K, N);
    }
    // ---- layer 2: 128 -> 256, WMMA ----
    {
        const int K = 128, N = 256;
        int n = N_NODES * K;
        zero_kernel<<<blocks(n, TB), TB, 0, stream>>>(agg, n);
        long long sthreads = (long long)N_EDGES * (K / 4);
        scatter_edges<<<blocks(sthreads, TB), TB, 0, stream>>>(x2, edges, agg, N_EDGES, K, K / 4, 4);
        long long ap = (long long)N_NODES * K / 2;
        pack_A_frag<<<blocks(ap, TB), TB, 0, stream>>>(x2,  Axp, N_NODES, K);
        pack_A_frag<<<blocks(ap, TB), TB, 0, stream>>>(agg, Agp, N_NODES, K);
        long long bp = (long long)K * N / 2;
        pack_B_frag<<<blocks(bp, TB), TB, 0, stream>>>(Ws2, Bsp, K, N);
        pack_B_frag<<<blocks(bp, TB), TB, 0, stream>>>(Wn2, Bnp, K, N);
        int waves = (N_NODES / 16) * (N / 64);           // 25000 waves
        wmma_gemm_packed<<<blocks((long long)waves * 32, TB), TB, 0, stream>>>(
            Axp, Agp, Bsp, Bnp, b2, x3, N_NODES, K, N);
    }
    // ---- head ----
    head_kernel<<<1, 256, 0, stream>>>(vs, bidx, x3, Wd, bd, Wo, bo, out);
    (void)in_sizes; (void)n_in; (void)out_size; (void)ws_size;
}
